// VectorQuantizer_17557826306285
// MI455X (gfx1250) — matrix operand order repaired and checked by reading, not compiled, not run on hardware
//
#include <hip/hip_runtime.h>
#include <hip/hip_bf16.h>
#include <math.h>

// ---------------------------------------------------------------------------
// VQ-VAE vector quantizer for MI455X (gfx1250, wave32, WMMA), fused version.
// z: [16,256,64,64] f32, embed: [1024,256] f32.
// Outputs (flat): loss(1), z_q NCHW(16777216), perplexity(1),
// min_encodings(67108864), idx(65536).
//
// Traffic budget @ 23.3 TB/s: z read 64MB (once, fused), z_q write 64MB (NT),
// one-hot write 268MB (NT), codebook from L2. ~400MB => ~17us floor.
// Distance GEMM = 34 GFLOP bf16 via v_wmma_f32_16x16x32_bf16 (negligible).
// ---------------------------------------------------------------------------

typedef __attribute__((ext_vector_type(16))) __bf16 v16bf;
typedef __attribute__((ext_vector_type(8)))  float  v8f;

#define N_TOK   65536
#define DIM     256
#define N_EMB   1024
#define ZQ_N    16777216
#define LDS_PAD 264   // 256 + 8 bf16 -> 528B row stride, rotates banks by 4

union FragBF {
    v16bf v;
    uint4 q[2];
    unsigned short s[16];
};

__device__ __forceinline__ unsigned short f32_to_bf16(float f) {
    unsigned u = __float_as_uint(f);
    unsigned r = u + 0x7FFFu + ((u >> 16) & 1u);   // round-to-nearest-even
    return (unsigned short)(r >> 16);
}

// ---------------------------------------------------------------------------
// Kernel 0: zero histogram + loss accumulator (per-call, deterministic)
// ---------------------------------------------------------------------------
__global__ __launch_bounds__(256) void vq_init(unsigned int* __restrict__ hist,
                                               float* __restrict__ loss_acc) {
    for (int k = threadIdx.x; k < N_EMB; k += 256) hist[k] = 0u;
    if (threadIdx.x == 0) *loss_acc = 0.0f;
}

// ---------------------------------------------------------------------------
// Kernel 1: embed f32 -> bf16 copy (B operand) + per-row |e|^2
// ---------------------------------------------------------------------------
__global__ __launch_bounds__(256) void vq_prep_embed(
    const float* __restrict__ embed, unsigned short* __restrict__ ebf,
    float* __restrict__ e2) {
    __shared__ float red[256];
    const int row = blockIdx.x;
    const int c   = threadIdx.x;
    float v = embed[(size_t)row * DIM + c];
    ebf[(size_t)row * DIM + c] = f32_to_bf16(v);
    red[c] = v * v;
    __syncthreads();
    for (int s = 128; s > 0; s >>= 1) {
        if (c < s) red[c] += red[c + s];
        __syncthreads();
    }
    if (c == 0) e2[row] = red[0];
}

// ---------------------------------------------------------------------------
// Kernel 2 (fused): WMMA argmin + gather + loss + one-hot + histogram.
// 16 tokens per block, 256 threads = 8 waves.
// Wave wv owns codewords [wv*128, wv*128+128) as 8 tiles of 16.
// score(n) = |e_n|^2 - 2 * <z_t, e_n>   (|z|^2 constant per token -> dropped)
// ---------------------------------------------------------------------------
__global__ __launch_bounds__(256) void vq_main_wmma(
    const float* __restrict__ z, const float* __restrict__ embed,
    const unsigned short* __restrict__ ebf, const float* __restrict__ e2,
    float* __restrict__ zq_out, float* __restrict__ minenc,
    float* __restrict__ idxf_out, unsigned int* __restrict__ hist,
    float* __restrict__ loss_acc) {
    __shared__ unsigned short zt[16 * LDS_PAD];  // 16 tokens x 256 dims bf16
    __shared__ float wb_val[8][16];
    __shared__ int   wb_idx[8][16];
    __shared__ int   lidx[16];
    __shared__ float red[256];

    const int tb = blockIdx.x * 16;       // token base
    const int b  = tb >> 12;
    const int h  = (tb >> 6) & 63;
    const int w0 = tb & 63;               // multiple of 16

    // ---- stage A-tile to LDS; keep f32 z values live for the gather phase --
    const int c = threadIdx.x;            // this thread's channel
    const size_t zoff = (((size_t)b * 256 + c) * 64 + h) * 64 + w0;
    union { float4 v4[4]; float f[16]; } zu;
    {
        const float4* gp = (const float4*)(z + zoff);
        zu.v4[0] = gp[0]; zu.v4[1] = gp[1]; zu.v4[2] = gp[2]; zu.v4[3] = gp[3];
        #pragma unroll
        for (int w = 0; w < 16; ++w)
            zt[w * LDS_PAD + c] = f32_to_bf16(zu.f[w]);
    }
    __syncthreads();

    const int lane = threadIdx.x & 31;
    const int wv   = threadIdx.x >> 5;
    const int hf   = lane >> 4;           // lane half (0/1)
    const int l16  = lane & 15;

    // ---- A fragments (ISA 16-bit A 16x32 layout), kept in registers --------
    // lanes 0-15 : halfs 0-7 = K 0..7,  halfs 8-15 = K 16..23
    // lanes 16-31: halfs 0-7 = K 8..15, halfs 8-15 = K 24..31   (M = lane%16)
    v16bf a[8];
    {
        const unsigned short* row = &zt[l16 * LDS_PAD];
        #pragma unroll
        for (int c8 = 0; c8 < 8; ++c8) {
            FragBF f;
            f.q[0] = *(const uint4*)(row + 32 * c8 + (hf ? 8 : 0));
            f.q[1] = *(const uint4*)(row + 32 * c8 + (hf ? 24 : 16));
            a[c8] = f.v;
        }
    }

    float bestVal[8];
    int   bestIdx[8];
    #pragma unroll
    for (int r = 0; r < 8; ++r) { bestVal[r] = 3.4e38f; bestIdx[r] = 0; }

    // ---- 8 codeword tiles of 16; preload all B frags, then 8 chained WMMAs -
    for (int t = 0; t < 8; ++t) {
        const int n = wv * 128 + t * 16 + l16;   // this lane's codeword column
        // B 32x16 layout: lane n holds K 0..7,16..23 of chunk, lane n+16 holds 8..15,24..31
        const unsigned short* brow = ebf + (size_t)n * DIM + (hf ? 8 : 0);
        FragBF bf[8];
        #pragma unroll
        for (int c8 = 0; c8 < 8; ++c8) {         // one clause of b128 loads
            bf[c8].q[0] = *(const uint4*)(brow + 32 * c8);
            bf[c8].q[1] = *(const uint4*)(brow + 32 * c8 + 16);
        }
        const float e2n = e2[n];
        v8f acc = {};
        #pragma unroll
        for (int c8 = 0; c8 < 8; ++c8)           // back-to-back WMMAs
            acc = __builtin_amdgcn_wmma_f32_16x16x32_bf16(
                false, a[c8], false, bf[c8].v, (short)0, acc, false, false);
        #pragma unroll
        for (int r = 0; r < 8; ++r) {    // r -> M = r (lanes<16) / 8+r (>=16)
            float s = e2n - 2.0f * acc[r];
            if (s < bestVal[r]) { bestVal[r] = s; bestIdx[r] = n; }
        }
    }

    // ---- reduce across 16 lanes of each half (same M rows) -----------------
    #pragma unroll
    for (int m = 1; m <= 8; m <<= 1) {
        #pragma unroll
        for (int r = 0; r < 8; ++r) {
            float ov = __shfl_xor(bestVal[r], m, 32);
            int   oi = __shfl_xor(bestIdx[r], m, 32);
            if (ov < bestVal[r] || (ov == bestVal[r] && oi < bestIdx[r])) {
                bestVal[r] = ov; bestIdx[r] = oi;
            }
        }
    }
    if (l16 == 0) {
        #pragma unroll
        for (int r = 0; r < 8; ++r) {
            wb_val[wv][hf * 8 + r] = bestVal[r];
            wb_idx[wv][hf * 8 + r] = bestIdx[r];
        }
    }
    __syncthreads();

    // ---- cross-wave reduction, one thread per token ------------------------
    if (threadIdx.x < 16) {
        const int m = threadIdx.x;
        float bv = wb_val[0][m];
        int   bi = wb_idx[0][m];
        #pragma unroll
        for (int w = 1; w < 8; ++w) {
            float v = wb_val[w][m];
            int   i = wb_idx[w][m];
            if (v < bv || (v == bv && i < bi)) { bv = v; bi = i; }
        }
        lidx[m]          = bi;
        idxf_out[tb + m] = (float)bi;
        atomicAdd(&hist[bi], 1u);
    }
    __syncthreads();

    // ---- gather phase: z_q, loss partial, one-hot --------------------------
    float part = 0.0f;
    #pragma unroll
    for (int w = 0; w < 16; ++w) {
        float e = embed[(size_t)lidx[w] * DIM + c];   // coalesced, L2-resident
        float d = e - zu.f[w];
        part += d * d;
        __builtin_nontemporal_store(e, &zq_out[zoff + w]);  // streaming, TH=NT
    }

    // one-hot rows: write all 1024 entries -> implicit zeroing of output
    #pragma unroll
    for (int w = 0; w < 16; ++w) {
        float* rowp = minenc + (size_t)(tb + w) * N_EMB;
        const int target = lidx[w];
        #pragma unroll
        for (int j = 0; j < 4; ++j) {
            int k = threadIdx.x + 256 * j;
            __builtin_nontemporal_store((k == target) ? 1.0f : 0.0f, &rowp[k]);
        }
    }

    red[threadIdx.x] = part;
    __syncthreads();
    for (int s = 128; s > 0; s >>= 1) {
        if (threadIdx.x < s) red[threadIdx.x] += red[threadIdx.x + s];
        __syncthreads();
    }
    if (threadIdx.x == 0) atomicAdd(loss_acc, red[0]);
}

// ---------------------------------------------------------------------------
// Kernel 3: scalars.  loss = 1.25 * sumsq / (B*C*H*W); perplexity from hist.
// ---------------------------------------------------------------------------
__global__ __launch_bounds__(256) void vq_finalize(
    const unsigned int* __restrict__ hist, const float* __restrict__ loss_acc,
    float* __restrict__ out_loss, float* __restrict__ out_perp) {
    __shared__ float red[256];
    float part = 0.0f;
    for (int k = threadIdx.x; k < N_EMB; k += 256) {
        float p = (float)hist[k] * (1.0f / (float)N_TOK);
        part += p * logf(p + 1e-10f);
    }
    red[threadIdx.x] = part;
    __syncthreads();
    for (int s = 128; s > 0; s >>= 1) {
        if (threadIdx.x < s) red[threadIdx.x] += red[threadIdx.x + s];
        __syncthreads();
    }
    if (threadIdx.x == 0) {
        *out_perp = expf(-red[0]);
        *out_loss = loss_acc[0] * 1.25f / (float)ZQ_N;
    }
}

// ---------------------------------------------------------------------------
extern "C" void kernel_launch(void* const* d_in, const int* in_sizes, int n_in,
                              void* d_out, int out_size, void* d_ws,
                              size_t ws_size, hipStream_t stream) {
    (void)in_sizes; (void)n_in; (void)out_size; (void)ws_size;
    const float* z     = (const float*)d_in[0];   // [16,256,64,64]
    const float* embed = (const float*)d_in[1];   // [1024,256]

    // workspace layout
    char* ws = (char*)d_ws;
    unsigned short* ebf      = (unsigned short*)(ws);            // 512 KB
    float*          e2       = (float*)(ws + 524288);            // 4 KB
    unsigned int*   hist     = (unsigned int*)(ws + 528384);     // 4 KB
    float*          loss_acc = (float*)(ws + 532480);            // 4 B

    // output layout (flat, reference return order)
    float* out       = (float*)d_out;
    float* out_loss  = out;                                  // [1]
    float* out_zq    = out + 1;                              // [16777216]
    float* out_perp  = out + 1 + (size_t)ZQ_N;               // [1]
    float* out_menc  = out + 2 + (size_t)ZQ_N;               // [67108864]
    float* out_idx   = out + 2 + (size_t)ZQ_N + (size_t)N_TOK * N_EMB; // [65536]

    vq_init<<<1, 256, 0, stream>>>(hist, loss_acc);
    vq_prep_embed<<<N_EMB, 256, 0, stream>>>(embed, ebf, e2);
    vq_main_wmma<<<N_TOK / 16, 256, 0, stream>>>(z, embed, ebf, e2, out_zq,
                                                 out_menc, out_idx, hist,
                                                 loss_acc);
    vq_finalize<<<1, 256, 0, stream>>>(hist, loss_acc, out_loss, out_perp);
}
